// SSMTransformerBlock_34041910788504
// MI455X (gfx1250) — matrix operand order repaired and checked by reading, not compiled, not run on hardware
//
#include <hip/hip_runtime.h>
#include <hip/hip_bf16.h>
#include <math.h>

// ---------------------------------------------------------------------------
// SSM transformer block for MI455X (gfx1250, wave32, WMMA + TDM).
// All large GEMMs run on v_wmma_f32_16x16x32_f16 (f16 in, f32 accumulate).
// K-slabs are staged into LDS by the Tensor Data Mover (tensor_load_to_lds)
// with double buffering so DMA overlaps WMMA; s_wait_tensorcnt + barrier
// provides readiness. Falls back to per-lane b128 staging if the TDM builtin
// is unavailable.
// ---------------------------------------------------------------------------

typedef _Float16 v16h __attribute__((ext_vector_type(16)));
typedef _Float16 v8h  __attribute__((ext_vector_type(8)));
typedef float    v8f  __attribute__((ext_vector_type(8)));
typedef unsigned int u32x4 __attribute__((ext_vector_type(4)));
typedef int          i32x4 __attribute__((ext_vector_type(4)));
typedef int          i32x8 __attribute__((ext_vector_type(8)));

#define LDS_ROW 40  // 32 halves per K-slab + 8 pad halves (bank-conflict pad)

#if defined(__HIP_DEVICE_COMPILE__) && __has_builtin(__builtin_amdgcn_tensor_load_to_lds) && __has_builtin(__builtin_amdgcn_s_wait_tensorcnt)
#define USE_TDM 1
#else
#define USE_TDM 0
#endif

#if USE_TDM
// ---------------------------------------------------------------------------
// Post one TDM descriptor: load a 128x32-half tile from a row-major f16
// tensor (leading dim `ld` elements) into LDS at byte offset `lds_off`,
// padding each 32-half row to LDS_ROW halves (pad 16 dwords data + 4 dwords).
// D# packing per CDNA5 ISA 8.3/8.4 (group0 128b, group1 256b; groups 2/3
// unused for 2D tiles -> zeros).  6-arg builtin form (clang-23 toolchain).
// ---------------------------------------------------------------------------
__device__ __forceinline__ void tdm_load_tile(unsigned lds_off,
                                              const _Float16* g, int ld)
{
  unsigned long long ga = (unsigned long long)(uintptr_t)g;
  u32x4 g0;
  g0.x = 1u;                                   // count=1, user descriptor
  g0.y = lds_off;                              // lds_addr (bytes)
  g0.z = (unsigned)ga;                         // global_addr[31:0]
  g0.w = (unsigned)((ga >> 32) & 0x01ffffffu)  // global_addr[56:32]
         | 0x80000000u;                        // type=2 ("image") in [127:126]
  i32x8 g1;
  g1[0] = (int)((1u << 16)                     // data_size = 2 bytes
                | (1u << 20)                   // pad_enable
                | (3u << 22)                   // pad_interval = 16 dwords
                | (3u << 25));                 // pad_amount   = 4 dwords
  g1[1] = (int)(32u << 16);                    // tensor_dim0 = 32 (lo16)
  g1[2] = (int)(128u << 16);                   // dim0 hi | tensor_dim1 = 128
  g1[3] = (int)(32u << 16);                    // dim1 hi | tile_dim0 = 32
  g1[4] = (int)128u;                           // tile_dim1 = 128, tile_dim2 = 0
  g1[5] = ld;                                  // tensor_dim0_stride[31:0]
  g1[6] = 0;                                   // stride hi | dim1_stride lo
  g1[7] = 0;
  i32x4 z4 = {};
  i32x8 z8 = {};
  __builtin_amdgcn_tensor_load_to_lds(g0, g1, z4, z4, z8, 0);
}
#endif

// ---------------------------------------------------------------------------
// Generic tiled WMMA GEMM:  out = epilogue( A[M,K] @ W^T[N,K] + bias )
//   A, W are f16; W is stored pre-transposed (row = N, contiguous K).
//   blockIdx.z = batch (per-head) index, advanced via element strides.
//   EPI: 0 = f16 out (+bias), 1 = f32 out (+bias), 2 = f32 out + residual,
//        3 = f16 out, exact GELU(+bias).
//   DUAL: accumulate a second (A2,W2) pair with the same K (Y = H C^T + U D^T).
// ---------------------------------------------------------------------------
template <int EPI, bool DUAL>
__global__ __launch_bounds__(256) void gemm_wmma(
    const _Float16* __restrict__ A, int lda, long aZ,
    const _Float16* __restrict__ W, int ldw, long wZ,
    const _Float16* __restrict__ A2, int lda2, long a2Z,
    const _Float16* __restrict__ W2, int ldw2, long w2Z,
    const float* __restrict__ bias,
    const float* __restrict__ resid,
    void* __restrict__ outp, int ldo, long oZ,
    int K)
{
  constexpr int NSTR = DUAL ? 2 : 1;      // input streams
#if USE_TDM
  constexpr int NBUF = 2;                 // ping-pong LDS buffers
#else
  constexpr int NBUF = 1;
#endif
  constexpr int TILE = 128 * LDS_ROW;     // halves per staged tile

  __shared__ __align__(16) _Float16 lA[NBUF * NSTR * TILE];
  __shared__ __align__(16) _Float16 lB[NBUF * NSTR * TILE];

  const int tid  = threadIdx.x;
  const int wv   = tid >> 5;       // wave id 0..7 -> owns 16 output rows
  const int lane = tid & 31;
  const int lr   = lane & 15;
  const int hl   = lane >> 4;      // lane half (ISA A/B fragment layout)
  const long mb  = (long)blockIdx.x * 128;
  const long nb  = (long)blockIdx.y * 128;
  const int  z   = blockIdx.z;

  A += (long)z * aZ;
  W += (long)z * wZ;
  if constexpr (DUAL) { A2 += (long)z * a2Z; W2 += (long)z * w2Z; }
  _Float16* outH = (_Float16*)outp + (long)z * oZ;
  float*    outF = (float*)outp + (long)z * oZ;

  const _Float16* gA[NSTR];
  const _Float16* gW[NSTR];
  int ldA[NSTR], ldW[NSTR];
  gA[0] = A; gW[0] = W; ldA[0] = lda; ldW[0] = ldw;
  if constexpr (DUAL) { gA[1] = A2; gW[1] = W2; ldA[1] = lda2; ldW[1] = ldw2; }

  v8f acc[8] = {};

  // ---- WMMA on one staged slab: fragment loads batched ahead of the MAC chain
  auto compute = [&](int buf) {
#pragma unroll
    for (int st = 0; st < NSTR; ++st) {
      const int tb = (buf * NSTR + st) * TILE;
      // A fragment (16x32): lane<16 K{0..7,16..23}, lane>=16 K{8..15,24..31}
      v8h alo = *(const v8h*)&lA[tb + (wv * 16 + lr) * LDS_ROW + hl * 8];
      v8h ahi = *(const v8h*)&lA[tb + (wv * 16 + lr) * LDS_ROW + hl * 8 + 16];
      v16h a = __builtin_shufflevector(alo, ahi, 0, 1, 2, 3, 4, 5, 6, 7, 8, 9, 10, 11, 12, 13, 14, 15);
      // B fragments (32x16): lane = column, lane<16 K{0..7,16..23}, lane>=16 K{8..15,24..31}
      v16h bf[8];
#pragma unroll
      for (int ct = 0; ct < 8; ++ct) {
        v8h blo = *(const v8h*)&lB[tb + (ct * 16 + lr) * LDS_ROW + hl * 8];
        v8h bhi = *(const v8h*)&lB[tb + (ct * 16 + lr) * LDS_ROW + hl * 8 + 16];
        bf[ct] = __builtin_shufflevector(blo, bhi, 0, 1, 2, 3, 4, 5, 6, 7, 8, 9, 10, 11, 12, 13, 14, 15);
      }
#pragma unroll
      for (int ct = 0; ct < 8; ++ct)
        acc[ct] = __builtin_amdgcn_wmma_f32_16x16x32_f16(
            false, a, false, bf[ct], (short)0, acc[ct], false, false);
    }
  };

#if USE_TDM
  // ---- double-buffered TDM pipeline: DMA slab s+1 while WMMAing slab s
  constexpr short NOPS = (short)(2 * NSTR);   // TDM ops per slab
  const unsigned ldsA0 = (unsigned)(uintptr_t)(void*)&lA[0];
  const unsigned ldsB0 = (unsigned)(uintptr_t)(void*)&lB[0];
  auto issue_slab = [&](int buf, int k0) {
#pragma unroll
    for (int st = 0; st < NSTR; ++st) {
      const unsigned off = (unsigned)((buf * NSTR + st) * TILE * 2);
      tdm_load_tile(ldsA0 + off, gA[st] + mb * ldA[st] + k0, ldA[st]);
      tdm_load_tile(ldsB0 + off, gW[st] + nb * ldW[st] + k0, ldW[st]);
    }
  };

  const int nSlab = K >> 5;
  if (wv == 0) issue_slab(0, 0);
  for (int s = 0; s < nSlab; ++s) {
    const int buf = s & 1;
    if (wv == 0) {
      if (s + 1 < nSlab) {
        issue_slab(buf ^ 1, (s + 1) * 32);
        __builtin_amdgcn_s_wait_tensorcnt(NOPS);   // slab s landed
      } else {
        __builtin_amdgcn_s_wait_tensorcnt((short)0);
      }
    }
    __syncthreads();          // slab s visible to all waves
    compute(buf);
    __syncthreads();          // all reads done before buf is DMA-overwritten
  }
#else
  // ---- fallback: per-lane b128 staging (2 threads per tile row)
  const int sr = tid >> 1;          // staging row 0..127
  const int sc = (tid & 1) * 16;    // staging col 0 or 16 (one uint4 copy = 8 halves = 16B)
  for (int k0 = 0; k0 < K; k0 += 32) {
#pragma unroll
    for (int st = 0; st < NSTR; ++st) {
      const int tb = st * TILE;
      *(uint4*)&lA[tb + sr * LDS_ROW + sc] =
          *(const uint4*)(gA[st] + (mb + sr) * ldA[st] + k0 + sc);
      *(uint4*)&lB[tb + sr * LDS_ROW + sc] =
          *(const uint4*)(gW[st] + (nb + sr) * ldW[st] + k0 + sc);
    }
    if (k0 + 32 < K) {
      __builtin_prefetch(gA[0] + (mb + sr) * ldA[0] + k0 + 32 + sc, 0, 1);
      __builtin_prefetch(gW[0] + (nb + sr) * ldW[0] + k0 + 32 + sc, 0, 1);
    }
    __syncthreads();
    compute(0);
    __syncthreads();
  }
#endif

  // ---- epilogue: C/D fragment layout: VGPR v, lanes 0-15 -> M=v, 16-31 -> M=8+v
#pragma unroll
  for (int ct = 0; ct < 8; ++ct) {
    const long col = nb + ct * 16 + lr;
    const float bv = bias ? bias[col] : 0.0f;
#pragma unroll
    for (int v = 0; v < 8; ++v) {
      const long row = mb + wv * 16 + hl * 8 + v;
      const long idx = row * (long)ldo + col;
      float val = acc[ct][v] + bv;
      if constexpr (EPI == 0) {
        outH[idx] = (_Float16)val;
      } else if constexpr (EPI == 1) {
        outF[idx] = val;
      } else if constexpr (EPI == 2) {
        outF[idx] = resid[idx] + val;
      } else {
        float g = 0.5f * val * (1.0f + erff(val * 0.70710678118654752f));
        outH[idx] = (_Float16)g;
      }
    }
  }
}

// ---------------------------------------------------------------------------
// LayerNorm over D=2048, one row per block, f32 in -> f16 out (GEMM operand).
// ---------------------------------------------------------------------------
__global__ __launch_bounds__(256) void ln_f16(
    const float* __restrict__ x, const float* __restrict__ w,
    const float* __restrict__ b, _Float16* __restrict__ out)
{
  const int D = 2048;
  const long row = blockIdx.x;
  const float* xr = x + row * D;
  const int tid = threadIdx.x;
  float v[8];
  float s = 0.0f, sq = 0.0f;
#pragma unroll
  for (int i = 0; i < 8; ++i) {
    float t = xr[i * 256 + tid];
    v[i] = t; s += t; sq += t * t;
  }
#pragma unroll
  for (int off = 16; off > 0; off >>= 1) {
    s  += __shfl_xor(s, off);
    sq += __shfl_xor(sq, off);
  }
  __shared__ float rs[8], rq[8], stats[2];
  if ((tid & 31) == 0) { rs[tid >> 5] = s; rq[tid >> 5] = sq; }
  __syncthreads();
  if (tid == 0) {
    float ts = 0.0f, tq = 0.0f;
    for (int i = 0; i < 8; ++i) { ts += rs[i]; tq += rq[i]; }
    float mean = ts / D;
    float var  = tq / D - mean * mean;
    stats[0] = mean;
    stats[1] = rsqrtf(var + 1e-5f);
  }
  __syncthreads();
  const float mean = stats[0], rstd = stats[1];
#pragma unroll
  for (int i = 0; i < 8; ++i) {
    const int c = i * 256 + tid;
    out[row * (long)D + c] = (_Float16)((v[i] - mean) * rstd * w[c] + b[c]);
  }
}

// ---------------------------------------------------------------------------
// f32 [R x C] -> f16 [C x R] (pre-transpose weights so GEMM W operand is K-major)
// ---------------------------------------------------------------------------
__global__ __launch_bounds__(256) void transpose_cvt(
    const float* __restrict__ src, _Float16* __restrict__ dst, int R, int C)
{
  __shared__ float tile[32][33];
  const int c0 = blockIdx.x * 32, r0 = blockIdx.y * 32;
  const int tx = threadIdx.x & 31, ty = threadIdx.x >> 5;
#pragma unroll
  for (int i = 0; i < 32; i += 8)
    tile[ty + i][tx] = src[(long)(r0 + ty + i) * C + c0 + tx];
  __syncthreads();
#pragma unroll
  for (int i = 0; i < 32; i += 8)
    dst[(long)(c0 + ty + i) * R + r0 + tx] = (_Float16)tile[tx][ty + i];
}

__global__ __launch_bounds__(256) void cvt_f16(
    const float* __restrict__ src, _Float16* __restrict__ dst, long n)
{
  long i = (long)blockIdx.x * 256 + threadIdx.x;
  if (i < n) dst[i] = (_Float16)src[i];
}

// ---------------------------------------------------------------------------
// Sequential SSM recurrence: h_t = A h_{t-1} + BU_t (BU precomputed by WMMA).
// Grid (head, batch); 256 threads: 2 threads per state element, A rows cached
// in VGPRs (64 f32 each), state in LDS, pair-reduce via shfl_xor.
// ---------------------------------------------------------------------------
__global__ __launch_bounds__(256) void ssm_scan(
    const float* __restrict__ Amat, const float* __restrict__ BU,
    _Float16* __restrict__ Hs, int T)
{
  const int h = blockIdx.x, b = blockIdx.y;
  const int tid = threadIdx.x;
  const int o = tid >> 1;     // state element 0..127
  const int k = tid & 1;      // which half of the 128-dot

  float areg[64];
  const float* arow = Amat + ((long)(h * 128 + o)) * 128 + k * 64;
#pragma unroll
  for (int j = 0; j < 64; ++j) areg[j] = arow[j];

  __shared__ __align__(16) float hs[128];
  if (tid < 128) hs[tid] = 0.0f;
  __syncthreads();

  const float* bu = BU + ((long)b * T) * 2048 + h * 128 + o;
  _Float16* hout  = Hs + ((long)b * T) * 2048 + h * 128 + o;

  for (int t = 0; t < T; ++t) {
    float p = 0.0f;
    const float4* hv = (const float4*)&hs[k * 64];
#pragma unroll
    for (int j = 0; j < 16; ++j) {
      float4 hh = hv[j];
      p += areg[4 * j + 0] * hh.x + areg[4 * j + 1] * hh.y +
           areg[4 * j + 2] * hh.z + areg[4 * j + 3] * hh.w;
    }
    p += __shfl_xor(p, 1);                       // combine the two half-dots
    const float nh = p + bu[(long)t * 2048];
    __syncthreads();                             // all reads of hs done
    if (k == 0) {
      hs[o] = nh;
      hout[(long)t * 2048] = (_Float16)nh;
    }
    __syncthreads();                             // new state visible
  }
}

// ---------------------------------------------------------------------------
extern "C" void kernel_launch(void* const* d_in, const int* in_sizes, int n_in,
                              void* d_out, int out_size, void* d_ws, size_t ws_size,
                              hipStream_t stream)
{
  (void)in_sizes; (void)n_in; (void)out_size; (void)ws_size;

  const float* x     = (const float*)d_in[0];
  const float* ln1w  = (const float*)d_in[1];
  const float* ln1b  = (const float*)d_in[2];
  const float* in_w  = (const float*)d_in[3];
  const float* in_b  = (const float*)d_in[4];
  const float* Amat  = (const float*)d_in[5];
  const float* Bm    = (const float*)d_in[6];
  const float* Cm    = (const float*)d_in[7];
  const float* Dm    = (const float*)d_in[8];
  const float* out_w = (const float*)d_in[9];
  const float* out_b = (const float*)d_in[10];
  const float* ln2w  = (const float*)d_in[11];
  const float* ln2b  = (const float*)d_in[12];
  const float* w1    = (const float*)d_in[13];
  const float* b1    = (const float*)d_in[14];
  const float* w2    = (const float*)d_in[15];
  const float* b2    = (const float*)d_in[16];
  float* out = (float*)d_out;

  const int D = 2048, HID = 8192, H = 16, T = 2048, Bb = 4;
  const long M = (long)Bb * T;                  // 8192 rows
  const size_t SZ_H = (size_t)M * D * 2;        // 32MB f16 activation
  const size_t SZ_F = (size_t)M * D * 4;        // 64MB f32 activation

  // Workspace regions with lifetime-based reuse (total ~194MB):
  char* ws = (char*)d_ws;
  _Float16* bufA = (_Float16*)(ws);                 // Xn -> Hs -> w1t
  _Float16* bufU = (_Float16*)(ws + SZ_H);          // U  -> w2t
  _Float16* bufY = (_Float16*)(ws + 2 * SZ_H);      // in_wt -> Y -> G chunk
  _Float16* bufM = (_Float16*)(ws + 3 * SZ_H);      // out_wt -> LN2 out
  float*    bufX = (float*)(ws + 4 * SZ_H);         // BU -> x2
  _Float16* bmh  = (_Float16*)(ws + 4 * SZ_H + SZ_F);
  _Float16* cmh  = bmh + H * 128 * 128;
  _Float16* dmh  = cmh + H * 128 * 128;

  const dim3 blk(256);
  const long nS = (long)H * 128 * 128;

  // ---- weight prep (f32 -> f16, transposed to K-major for the W operand)
  transpose_cvt<<<dim3(D / 32, D / 32), blk, 0, stream>>>(in_w, bufY, D, D);
  transpose_cvt<<<dim3(D / 32, D / 32), blk, 0, stream>>>(out_w, bufM, D, D);
  cvt_f16<<<(nS + 255) / 256, blk, 0, stream>>>(Bm, bmh, nS);
  cvt_f16<<<(nS + 255) / 256, blk, 0, stream>>>(Cm, cmh, nS);
  cvt_f16<<<(nS + 255) / 256, blk, 0, stream>>>(Dm, dmh, nS);

  // ---- LN1 -> Xn (f16)
  ln_f16<<<M, blk, 0, stream>>>(x, ln1w, ln1b, bufA);

  // ---- U = Xn @ in_w + in_b  (f16 out)
  gemm_wmma<0, false><<<dim3(64, 16, 1), blk, 0, stream>>>(
      bufA, D, 0, bufY, D, 0, nullptr, 0, 0, nullptr, 0, 0,
      in_b, nullptr, bufU, D, 0, D);

  // ---- BU = U @ Bm^T per head (f32 out, feeds recurrence)
  gemm_wmma<1, false><<<dim3(64, 1, H), blk, 0, stream>>>(
      bufU, D, 128, bmh, 128, 128 * 128, nullptr, 0, 0, nullptr, 0, 0,
      nullptr, nullptr, bufX, D, 128, 128);

  // ---- recurrence: h_t = A h_{t-1} + BU_t, store all states as f16
  ssm_scan<<<dim3(H, Bb), blk, 0, stream>>>(Amat, bufX, bufA, T);

  // ---- Y = Hs @ Cm^T + U @ Dm^T per head (dual-stream WMMA, f16 out)
  gemm_wmma<0, true><<<dim3(64, 1, H), blk, 0, stream>>>(
      bufA, D, 128, cmh, 128, 128 * 128,
      bufU, D, 128, dmh, 128, 128 * 128,
      nullptr, nullptr, bufY, D, 128, 128);

  // ---- x2 = x + Y @ out_w + out_b (f32 out)
  gemm_wmma<2, false><<<dim3(64, 16, 1), blk, 0, stream>>>(
      bufY, D, 0, bufM, D, 0, nullptr, 0, 0, nullptr, 0, 0,
      out_b, x, bufX, D, 0, D);

  // ---- LN2 -> f16
  ln_f16<<<M, blk, 0, stream>>>(bufX, ln2w, ln2b, bufM);

  // ---- MLP weights (transposed into regions freed above)
  transpose_cvt<<<dim3(HID / 32, D / 32), blk, 0, stream>>>(w1, bufA, D, HID);
  transpose_cvt<<<dim3(D / 32, HID / 32), blk, 0, stream>>>(w2, bufU, HID, D);

  // ---- MLP in 4 row-chunks of 2048 (G chunk reuses the 32MB Y region)
  for (int c = 0; c < 4; ++c) {
    const long ro = (long)c * 2048;
    gemm_wmma<3, false><<<dim3(16, 64, 1), blk, 0, stream>>>(
        bufM + ro * D, D, 0, bufA, D, 0, nullptr, 0, 0, nullptr, 0, 0,
        b1, nullptr, bufY, HID, 0, D);
    gemm_wmma<2, false><<<dim3(16, 16, 1), blk, 0, stream>>>(
        bufY, HID, 0, bufU, HID, 0, nullptr, 0, 0, nullptr, 0, 0,
        b2, bufX + ro * D, out + ro * D, D, 0, HID);
  }
}
